// Compressor_45140106281443
// MI455X (gfx1250) — matrix operand order repaired and checked by reading, not compiled, not run on hardware
//
#include <hip/hip_runtime.h>
#include <hip/hip_bf16.h>

// ---------------------------------------------------------------------------
// MoE compressor for MI455X (gfx1250, wave32).
//  K0a: x fp32 -> f16 copy                     (xh  [T, D])
//  K0b: compress_neurons fp32 -> f16 TRANSPOSED (cnh [N, R, D])
//  K1 : fp32 router + top-8 + softmax + per-neuron dispatch lists
//  K2 : grouped GEMM per neuron on v_wmma_f32_16x16x32_f16; A/B tiles staged
//       via global_load_async_to_lds_b128 (ASYNCcnt); operands are pure
//       ds_load_b128 reads matching the ISA 16-bit WMMA layouts exactly.
//  K3 : deterministic 8-way reduce into the output.
// ---------------------------------------------------------------------------

typedef __attribute__((ext_vector_type(16))) _Float16 v16h;
typedef __attribute__((ext_vector_type(8)))  _Float16 v8h;
typedef __attribute__((ext_vector_type(4)))  _Float16 v4h;
typedef __attribute__((ext_vector_type(8)))  float    v8f;
typedef __attribute__((ext_vector_type(4)))  int      v4i;

#define T_TOK   8192        // B*S
#define D_DIM   1024
#define R_DIM   128
#define N_EXP   64
#define TOPK    8
#define KCH     32          // K-chunk matching wmma f32_16x16x32_f16
#define TILE_M  16

// d_out regions (float element offsets), reference return order:
// output [T,R], weights [T,K], topk_idx [T,K] (as float), scores [T,N]
#define O_OUT   0
#define O_W     (T_TOK * R_DIM)
#define O_IDX   (O_W + T_TOK * TOPK)
#define O_SC    (O_IDX + T_TOK * TOPK)

// workspace layout (byte offsets), all 256-aligned
#define WS_COUNTS 0u
#define WS_LIST   256u
#define WS_PROJ   (WS_LIST + (size_t)N_EXP * T_TOK * 4)        // [T*K, R] f32
#define WS_XH     (WS_PROJ + (size_t)T_TOK * TOPK * R_DIM * 4) // [T, D]  f16
#define WS_CNH    (WS_XH   + (size_t)T_TOK * D_DIM * 2)        // [N, R, D] f16

#if __has_builtin(__builtin_amdgcn_global_load_async_to_lds_b128) && \
    __has_builtin(__builtin_amdgcn_s_wait_asynccnt)
#define HAVE_ASYNC_LDS 1
typedef __attribute__((address_space(1))) v4i* gv4i_p;   // global (AS1) v4i*
typedef __attribute__((address_space(3))) v4i* lv4i_p;   // LDS (AS3, 32-bit) v4i*
#endif

// ---------------------------------------------------------------------------
// K0a: x fp32 -> f16 (elementwise, float4 in / 4xf16 out)
// ---------------------------------------------------------------------------
__global__ __launch_bounds__(256)
void cvt_x_kernel(const float* __restrict__ x, _Float16* __restrict__ xh)
{
    const size_t i = ((size_t)blockIdx.x * 256 + threadIdx.x) * 4;
    float4 v = *(const float4*)(x + i);
    v4h h; h[0] = (_Float16)v.x; h[1] = (_Float16)v.y;
           h[2] = (_Float16)v.z; h[3] = (_Float16)v.w;
    *(v4h*)(xh + i) = h;
}

// ---------------------------------------------------------------------------
// K0b: cn [N,D,R] fp32 -> cnh [N,R,D] f16 (LDS-tiled 32x32 transpose)
// grid (N, D/32, R/32)
// ---------------------------------------------------------------------------
__global__ __launch_bounds__(256)
void cvt_w_kernel(const float* __restrict__ cn, _Float16* __restrict__ cnh)
{
    __shared__ float tile[32][33];                 // +1 pad: no bank conflicts
    const int n  = blockIdx.x;
    const int db = blockIdx.y * 32;
    const int rb = blockIdx.z * 32;
    const int tid = threadIdx.x;
    #pragma unroll
    for (int it = 0; it < 4; ++it) {
        const int idx = tid + it * 256;
        const int dr = idx >> 5, rc = idx & 31;    // read coalesced along r
        tile[dr][rc] = cn[((size_t)n * D_DIM + db + dr) * R_DIM + rb + rc];
    }
    __syncthreads();
    #pragma unroll
    for (int it = 0; it < 4; ++it) {
        const int idx = tid + it * 256;
        const int rr = idx >> 5, dc = idx & 31;    // write coalesced along d
        cnh[((size_t)n * R_DIM + rb + rr) * D_DIM + db + dc] = (_Float16)tile[dc][rr];
    }
}

// ---------------------------------------------------------------------------
// K1: router scores (exact fp32 so top-k matches reference), top-8,
// softmax, dispatch-list build. One 64-thread block (2 waves) per token.
// ---------------------------------------------------------------------------
__global__ __launch_bounds__(N_EXP)
void router_topk_kernel(const float* __restrict__ x,
                        const float* __restrict__ rw,
                        float* __restrict__ out,
                        int* __restrict__ counts,
                        int* __restrict__ dlist)
{
    __shared__ float s_sc[N_EXP];
    const int t = blockIdx.x;
    const int n = threadIdx.x;                    // 0..63, one neuron each

    const float4* xr = (const float4*)(x + (size_t)t * D_DIM);
    const float4* wr = (const float4*)(rw + (size_t)n * D_DIM);
    float acc = 0.f;
    #pragma unroll 4
    for (int d = 0; d < D_DIM / 4; ++d) {
        float4 a = xr[d], b = wr[d];
        acc += a.x * b.x + a.y * b.y + a.z * b.z + a.w * b.w;
    }
    s_sc[n] = acc;
    out[O_SC + (size_t)t * N_EXP + n] = acc;      // scores output
    __syncthreads();

    if (n == 0) {
        int   idxs[TOPK];
        float vals[TOPK];
        for (int k = 0; k < TOPK; ++k) {          // ties -> lowest index
            int bi = 0; float bv = s_sc[0];
            for (int i = 1; i < N_EXP; ++i) {
                float v = s_sc[i];
                if (v > bv) { bv = v; bi = i; }
            }
            idxs[k] = bi; vals[k] = bv;
            s_sc[bi] = -3.402823466e38f;
        }
        float m = vals[0], e[TOPK], s = 0.f;
        #pragma unroll
        for (int k = 0; k < TOPK; ++k) { e[k] = __expf(vals[k] - m); s += e[k]; }
        const float inv = 1.f / s;
        #pragma unroll
        for (int k = 0; k < TOPK; ++k) {
            const float w = e[k] * inv;
            out[O_W   + t * TOPK + k] = w;
            out[O_IDX + t * TOPK + k] = (float)idxs[k];
            const int pos = atomicAdd(&counts[idxs[k]], 1);
            dlist[idxs[k] * T_TOK + pos] = t * TOPK + k;   // entry = token*8 + slot
        }
    }
}

// ---------------------------------------------------------------------------
// K2: grouped projection GEMM. Block = 8 wave32s (256 threads).
// blockIdx.x = neuron n, blockIdx.y = 16-token tile of n's dispatch list.
// Wave j computes the 16x16 C tile for output columns [16j, 16j+16).
// ---------------------------------------------------------------------------
__global__ __launch_bounds__(256)
void moe_proj_kernel(const _Float16* __restrict__ xh,   // [T, D]
                     const _Float16* __restrict__ cnh,  // [N, R, D]
                     const float* __restrict__ outbuf,  // weights region read
                     const int* __restrict__ counts,
                     const int* __restrict__ dlist,
                     float* __restrict__ proj)          // [T*K, R]
{
    __shared__ int s_ids[TILE_M];
    __shared__ __attribute__((aligned(16))) _Float16 s_a[TILE_M * KCH]; // 16x32 f16 (1 KB)
    __shared__ __attribute__((aligned(16))) _Float16 s_b[R_DIM * KCH];  // 128x32 f16 (8 KB)

    const int n    = blockIdx.x;
    const int base = blockIdx.y * TILE_M;
    const int cnt  = counts[n];
    if (base >= cnt) return;               // uniform early-exit

    const int tid  = threadIdx.x;
    const int lane = tid & 31;
    const int wv   = tid >> 5;             // 0..7 -> 16-wide output column block
    const int mr   = lane & 15;            // A row / B,C column within tile
    const int klo  = (lane < 16) ? 0 : 8;  // A operand K base (ISA 7.12.2)
    const int kb0  = (lane < 16) ? 0 : 8;  // B operand K base (ISA 7.12.4 dense analog)

    if (tid < TILE_M) {
        int src = base + tid;
        if (src >= cnt) src = base;        // pad: duplicate row 0 (benign rewrite)
        s_ids[tid] = dlist[n * T_TOK + src];
    }

#ifdef HAVE_ASYNC_LDS
    const unsigned lA = (unsigned)(__UINTPTR_TYPE__)&s_a[0];
    const unsigned lB = (unsigned)(__UINTPTR_TYPE__)&s_b[0];
#endif

    v8f acc = {};
    for (int d0 = 0; d0 < D_DIM; d0 += KCH) {
        __syncthreads();                   // prior compute done / ids visible

#ifdef HAVE_ASYNC_LDS
        // A tile: 16 token rows x 32 halves = 64 B/row -> 64 b128 transfers
        if (tid < 64) {
            const int row = tid >> 2, seg = tid & 3;
            const int tok = s_ids[row] >> 3;               // entry -> token
            const char* gsa = (const char*)xh + ((size_t)tok * D_DIM + d0) * 2 + seg * 16;
            __builtin_amdgcn_global_load_async_to_lds_b128(
                (gv4i_p)(__UINTPTR_TYPE__)gsa,
                (lv4i_p)(lA + (unsigned)(row * 64 + seg * 16)), 0, 0);
        }
        // B tile: 128 r rows x 32 halves = 64 B/row -> 512 b128 transfers
        #pragma unroll
        for (int it = 0; it < 2; ++it) {
            const int idx = tid + it * 256;
            const int r = idx >> 2, seg = idx & 3;
            const char* gsb = (const char*)cnh +
                (((size_t)n * R_DIM + r) * D_DIM + d0) * 2 + seg * 16;
            __builtin_amdgcn_global_load_async_to_lds_b128(
                (gv4i_p)(__UINTPTR_TYPE__)gsb,
                (lv4i_p)(lB + (unsigned)(r * 64 + seg * 16)), 0, 0);
        }
        __builtin_amdgcn_s_wait_asynccnt(0);
#else
        if (tid < 64) {
            const int row = tid >> 2, seg = tid & 3;
            const int tok = s_ids[row] >> 3;
            *(float4*)((char*)s_a + row * 64 + seg * 16) =
                *(const float4*)((const char*)xh + ((size_t)tok * D_DIM + d0) * 2 + seg * 16);
        }
        #pragma unroll
        for (int it = 0; it < 2; ++it) {
            const int idx = tid + it * 256;
            const int r = idx >> 2, seg = idx & 3;
            *(float4*)((char*)s_b + r * 64 + seg * 16) =
                *(const float4*)((const char*)cnh +
                    (((size_t)n * R_DIM + r) * D_DIM + d0) * 2 + seg * 16);
        }
#endif
        __syncthreads();

        // ---- operands: pure b128 LDS reads matching ISA 16-bit layouts
        // A lane: row mr, K = {klo+0..7, klo+16..23} -> two contiguous 16B runs
        const _Float16* pa = &s_a[mr * KCH + klo];
        v8h a0 = *(const v8h*)pa;
        v8h a1 = *(const v8h*)(pa + 16);
        v16h a = __builtin_shufflevector(a0, a1,
                 0,1,2,3,4,5,6,7,8,9,10,11,12,13,14,15);
        // B lane: column (wv*16+mr), K = {kb0+0..7, kb0+16..23} -> two contiguous 16B runs
        const _Float16* pb = &s_b[(wv * 16 + mr) * KCH + kb0];
        v8h b0 = *(const v8h*)pb;
        v8h b1 = *(const v8h*)(pb + 16);
        v16h b = __builtin_shufflevector(b0, b1,
                 0,1,2,3,4,5,6,7,8,9,10,11,12,13,14,15);

        acc = __builtin_amdgcn_wmma_f32_16x16x32_f16(
                  false, a, false, b, (short)0, acc, false, false);
    }

    // ---- write weighted tile: C layout = VGPR p holds row (p + 8*(lane>=16))
    #pragma unroll
    for (int p = 0; p < 8; ++p) {
        const int m     = p + ((lane < 16) ? 0 : 8);
        const int entry = s_ids[m];
        const float wgt = outbuf[O_W + entry];               // softmax weight
        proj[(size_t)entry * R_DIM + wv * 16 + mr] = acc[p] * wgt;
    }
}

// ---------------------------------------------------------------------------
// K3: output[t, r] = sum_k proj[t*8+k, r]  (deterministic reduce)
// ---------------------------------------------------------------------------
__global__ __launch_bounds__(R_DIM)
void reduce_kernel(const float* __restrict__ proj, float* __restrict__ out)
{
    const int t = blockIdx.x;
    const int r = threadIdx.x;
    float s = 0.f;
    #pragma unroll
    for (int k = 0; k < TOPK; ++k)
        s += proj[((size_t)t * TOPK + k) * R_DIM + r];
    out[O_OUT + (size_t)t * R_DIM + r] = s;
}

// ---------------------------------------------------------------------------
extern "C" void kernel_launch(void* const* d_in, const int* in_sizes, int n_in,
                              void* d_out, int out_size, void* d_ws, size_t ws_size,
                              hipStream_t stream) {
    const float* x  = (const float*)d_in[0];   // [4,2048,1024]
    const float* rw = (const float*)d_in[1];   // [64,1024]
    const float* cn = (const float*)d_in[2];   // [64,1024,128]
    float* out = (float*)d_out;

    char*      ws     = (char*)d_ws;
    int*       counts = (int*)(ws + WS_COUNTS);
    int*       dlist  = (int*)(ws + WS_LIST);
    float*     proj   = (float*)(ws + WS_PROJ);
    _Float16*  xh     = (_Float16*)(ws + WS_XH);
    _Float16*  cnh    = (_Float16*)(ws + WS_CNH);

    (void)hipMemsetAsync(counts, 0, N_EXP * sizeof(int), stream);

    cvt_x_kernel<<<(T_TOK * D_DIM) / (256 * 4), 256, 0, stream>>>(x, xh);

    dim3 gw(N_EXP, D_DIM / 32, R_DIM / 32);
    cvt_w_kernel<<<gw, 256, 0, stream>>>(cn, cnh);

    router_topk_kernel<<<T_TOK, N_EXP, 0, stream>>>(x, rw, out, counts, dlist);

    dim3 g2(N_EXP, T_TOK / TILE_M);
    moe_proj_kernel<<<g2, 256, 0, stream>>>(xh, cnh, out, counts, dlist, proj);

    reduce_kernel<<<T_TOK, R_DIM, 0, stream>>>(proj, out);
}
